// _SelfAttention_71322226917566
// MI455X (gfx1250) — compile-verified, matrix operand order checked by reading, not run on hardware
//
#include <hip/hip_runtime.h>
#include <hip/hip_bf16.h>

// ---------------------------------------------------------------------------
// Self-attention (B=4, S=2048, H=1024) for MI455X / gfx1250.
// All 6 matmuls run on v_wmma_f32_16x16x32_bf16 (fp32->bf16 inputs, f32 acc).
// GEMM: block tile 128x128 (8 waves), wave tile 64x32 = 4x2 WMMA tiles.
// K is a compile-time constant (1024 or 2048) so all six row addresses fold
// into 24-bit instruction offsets off TWO base pointers -> no address VALU in
// the loop, no WMMA->VALU hazard NOPs, clean clause+staggered-wait pipeline.
// ---------------------------------------------------------------------------

typedef __attribute__((ext_vector_type(16))) __bf16        v16bf;
typedef __attribute__((ext_vector_type(8)))  float         v8f;
typedef __attribute__((ext_vector_type(8)))  unsigned int  v8u;

#define H_DIM   1024
#define S_LEN   2048
#define B_SZ    4
#define M_ROWS  (B_SZ * S_LEN)   // 8192

// ---- fp32 -> bf16 (round to nearest even) ----------------------------------
__device__ __forceinline__ unsigned short f2bf(float f) {
    unsigned int u = __builtin_bit_cast(unsigned int, f);
    u += 0x7FFFu + ((u >> 16) & 1u);
    return (unsigned short)(u >> 16);
}

__global__ void f32_to_bf16_kernel(const float* __restrict__ in,
                                   unsigned short* __restrict__ out, int n4) {
    int i = blockIdx.x * blockDim.x + threadIdx.x;
    if (i < n4) {
        float4 f = reinterpret_cast<const float4*>(in)[i];
        reinterpret_cast<ushort4*>(out)[i] =
            make_ushort4(f2bf(f.x), f2bf(f.y), f2bf(f.z), f2bf(f.w));
    }
}

// ---- bf16 fragment load (16x32 A layout; lanes 16..31 take K+8/K+24 half) ---
__device__ __forceinline__ v16bf ldfrag(const unsigned short* __restrict__ p) {
    uint4 lo = *reinterpret_cast<const uint4*>(p);        // K .. K+7
    uint4 hi = *reinterpret_cast<const uint4*>(p + 16);   // K+16 .. K+23
    v8u u = { lo.x, lo.y, lo.z, lo.w, hi.x, hi.y, hi.z, hi.w };
    return __builtin_bit_cast(v16bf, u);
}

__device__ __forceinline__ v8f bfwmma(v16bf a, v16bf b, v8f c) {
    return __builtin_amdgcn_wmma_f32_16x16x32_bf16(
        /*neg_a=*/false, a, /*neg_b=*/false, b,
        /*c_mod=*/(short)0, c, /*reuse_a=*/false, /*reuse_b=*/false);
}

// ---- generic C[m,n] = scale * sum_k A[m,k]*Bt[n,k] (+bias[n]) --------------
// Block: 256 threads = 8 waves (2 M x 4 N); block tile 128x128; wave tile 64x32.
// KC (compile-time) is the K extent and the row stride of A and Bt.
template <bool OUT_BF16, int KC>
__global__ void __launch_bounds__(256, 1)
gemm_abt_kernel(const unsigned short* __restrict__ A,
                const unsigned short* __restrict__ Bt,
                const float* __restrict__ bias,
                void* __restrict__ Cout,
                int M, int N, float scale,
                long long strideA, long long strideB,
                long long strideC) {
    const int lane = threadIdx.x & 31;
    const int wave = threadIdx.x >> 5;
    const int wm   = (wave & 1) << 6;   // 0,64
    const int wn   = (wave >> 1) << 5;  // 0,32,64,96
    const long long bz = blockIdx.z;

    const int bm = blockIdx.y * 128 + wm;
    const int bn = blockIdx.x * 128 + wn;
    const int r  = lane & 15;
    const int kh = (lane & 16) ? 8 : 0;  // lane-half K offset

    // Two base pointers; all row offsets are compile-time immediates (mt*16*KC).
    const unsigned short* pa = A  + bz * strideA + (size_t)(bm + r) * KC + kh;
    const unsigned short* pb = Bt + bz * strideB + (size_t)(bn + r) * KC + kh;

    v8f acc[4][2];
#pragma unroll
    for (int mt = 0; mt < 4; ++mt)
#pragma unroll
        for (int nt = 0; nt < 2; ++nt)
            acc[mt][nt] = (v8f){};

#pragma unroll 4
    for (int k0 = 0; k0 < KC; k0 += 32) {
        v16bf b0 = ldfrag(pb + 0 * 16 * KC + k0);
        v16bf b1 = ldfrag(pb + 1 * 16 * KC + k0);
        v16bf a0 = ldfrag(pa + 0 * 16 * KC + k0);
        v16bf a1 = ldfrag(pa + 1 * 16 * KC + k0);
        v16bf a2 = ldfrag(pa + 2 * 16 * KC + k0);
        v16bf a3 = ldfrag(pa + 3 * 16 * KC + k0);
        acc[0][0] = bfwmma(a0, b0, acc[0][0]);
        acc[0][1] = bfwmma(a0, b1, acc[0][1]);
        acc[1][0] = bfwmma(a1, b0, acc[1][0]);
        acc[1][1] = bfwmma(a1, b1, acc[1][1]);
        acc[2][0] = bfwmma(a2, b0, acc[2][0]);
        acc[2][1] = bfwmma(a2, b1, acc[2][1]);
        acc[3][0] = bfwmma(a3, b0, acc[3][0]);
        acc[3][1] = bfwmma(a3, b1, acc[3][1]);
    }

    // Epilogue: C/D layout — VGPR r8, lane L: row = r8 + 8*(L>=16), col = L%16.
    const int rowadd = (lane >> 4) * 8;
#pragma unroll
    for (int mt = 0; mt < 4; ++mt) {
#pragma unroll
        for (int nt = 0; nt < 2; ++nt) {
            const int col = bn + nt * 16 + (lane & 15);
            const float bvv = bias ? bias[col] : 0.0f;
#pragma unroll
            for (int r8 = 0; r8 < 8; ++r8) {
                const int row = bm + mt * 16 + rowadd + r8;
                const float val = acc[mt][nt][r8] * scale + bvv;
                const long long idx = bz * strideC + (long long)row * N + col;
                if (OUT_BF16)
                    reinterpret_cast<unsigned short*>(Cout)[idx] = f2bf(val);
                else
                    reinterpret_cast<float*>(Cout)[idx] = val;
            }
        }
    }
}

// ---- bf16 transpose [S,H] -> [H,S] per batch (64x64 LDS tiles) -------------
__global__ void transpose_bf16_kernel(const unsigned short* __restrict__ in,
                                      unsigned short* __restrict__ out) {
    __shared__ unsigned short tile[64][65];
    const int b  = blockIdx.z;
    const unsigned short* src = in  + (size_t)b * S_LEN * H_DIM;
    unsigned short*       dst = out + (size_t)b * S_LEN * H_DIM;
    const int e0 = blockIdx.x * 64, s0 = blockIdx.y * 64;
    const int tx = threadIdx.x & 63, r0 = threadIdx.x >> 6;
#pragma unroll
    for (int i = 0; i < 16; ++i) {
        const int row = r0 + i * 4;  // s within tile
        tile[row][tx] = src[(size_t)(s0 + row) * H_DIM + e0 + tx];
    }
    __syncthreads();
#pragma unroll
    for (int i = 0; i < 16; ++i) {
        const int row = r0 + i * 4;  // e within tile
        dst[(size_t)(e0 + row) * S_LEN + s0 + tx] = tile[tx][row];
    }
}

// ---- row softmax over 2048 cols: one wave32 per row, bf16 output -----------
__global__ void softmax_rows_kernel(const float* __restrict__ scores,
                                    unsigned short* __restrict__ p) {
    const int wave = threadIdx.x >> 5;
    const int lane = threadIdx.x & 31;
    const long long row = (long long)blockIdx.x * 8 + wave;
    const float* src = scores + row * S_LEN;

    float4 v[16];
    float m = -3.402823466e38f;
#pragma unroll
    for (int j = 0; j < 16; ++j) {
        v[j] = reinterpret_cast<const float4*>(src)[j * 32 + lane];
        m = fmaxf(m, fmaxf(fmaxf(v[j].x, v[j].y), fmaxf(v[j].z, v[j].w)));
    }
#pragma unroll
    for (int off = 16; off >= 1; off >>= 1)
        m = fmaxf(m, __shfl_xor(m, off, 32));

    float s = 0.0f;
#pragma unroll
    for (int j = 0; j < 16; ++j) {
        v[j].x = __expf(v[j].x - m); v[j].y = __expf(v[j].y - m);
        v[j].z = __expf(v[j].z - m); v[j].w = __expf(v[j].w - m);
        s += v[j].x + v[j].y + v[j].z + v[j].w;
    }
#pragma unroll
    for (int off = 16; off >= 1; off >>= 1)
        s += __shfl_xor(s, off, 32);

    const float inv = 1.0f / s;
    unsigned short* prow = p + row * S_LEN;
#pragma unroll
    for (int j = 0; j < 16; ++j) {
        reinterpret_cast<ushort4*>(prow)[j * 32 + lane] =
            make_ushort4(f2bf(v[j].x * inv), f2bf(v[j].y * inv),
                         f2bf(v[j].z * inv), f2bf(v[j].w * inv));
    }
}

// ---------------------------------------------------------------------------
extern "C" void kernel_launch(void* const* d_in, const int* in_sizes, int n_in,
                              void* d_out, int out_size, void* d_ws, size_t ws_size,
                              hipStream_t stream) {
    (void)in_sizes; (void)n_in; (void)out_size; (void)ws_size;

    const float* x  = (const float*)d_in[0];
    const float* Wq = (const float*)d_in[1];
    const float* bq = (const float*)d_in[2];
    const float* Wk = (const float*)d_in[3];
    const float* bk = (const float*)d_in[4];
    const float* Wv = (const float*)d_in[5];
    const float* bv = (const float*)d_in[6];
    const float* Wo = (const float*)d_in[7];
    const float* bo = (const float*)d_in[8];
    float* out = (float*)d_out;

    // Workspace layout (bytes)
    unsigned char* ws = (unsigned char*)d_ws;
    const size_t XB   = (size_t)M_ROWS * H_DIM * 2;   // 16 MB
    const size_t WB   = (size_t)H_DIM * H_DIM * 2;    //  2 MB each
    const size_t QB   = XB;                            // 16 MB each
    const size_t SCR  = (size_t)B_SZ * S_LEN * S_LEN * 4;  // 64 MB
    unsigned short* xb  = (unsigned short*)(ws);               // also vT after QKV
    unsigned short* wqb = (unsigned short*)(ws + XB);
    unsigned short* wkb = (unsigned short*)(ws + XB + WB);
    unsigned short* wvb = (unsigned short*)(ws + XB + 2 * WB);
    unsigned short* wob = (unsigned short*)(ws + XB + 3 * WB);
    unsigned short* qb  = (unsigned short*)(ws + XB + 4 * WB); // also attn-out
    unsigned short* kb  = (unsigned short*)(ws + XB + 4 * WB + QB);
    unsigned short* vb  = (unsigned short*)(ws + XB + 4 * WB + 2 * QB);
    float*          sc  = (float*)(ws + XB + 4 * WB + 3 * QB);
    unsigned short* pb  = (unsigned short*)(ws + XB + 4 * WB + 3 * QB + SCR);
    unsigned short* vtb = xb;   // x no longer needed after QKV GEMMs
    unsigned short* ab  = qb;   // Q no longer needed after scores GEMM

    dim3 blk(256);

    // 1) fp32 -> bf16 conversions
    {
        int n4 = M_ROWS * H_DIM / 4;
        f32_to_bf16_kernel<<<dim3((n4 + 255) / 256), blk, 0, stream>>>(x, xb, n4);
        int w4 = H_DIM * H_DIM / 4;
        dim3 g((w4 + 255) / 256);
        f32_to_bf16_kernel<<<g, blk, 0, stream>>>(Wq, wqb, w4);
        f32_to_bf16_kernel<<<g, blk, 0, stream>>>(Wk, wkb, w4);
        f32_to_bf16_kernel<<<g, blk, 0, stream>>>(Wv, wvb, w4);
        f32_to_bf16_kernel<<<g, blk, 0, stream>>>(Wo, wob, w4);
    }

    // 2) Q/K/V = x @ W^T + b   (M=8192, N=K=1024), bf16 out
    {
        dim3 g(H_DIM / 128, M_ROWS / 128, 1);
        gemm_abt_kernel<true, H_DIM><<<g, blk, 0, stream>>>(xb, wqb, bq, qb,
            M_ROWS, H_DIM, 1.0f, 0, 0, 0);
        gemm_abt_kernel<true, H_DIM><<<g, blk, 0, stream>>>(xb, wkb, bk, kb,
            M_ROWS, H_DIM, 1.0f, 0, 0, 0);
        gemm_abt_kernel<true, H_DIM><<<g, blk, 0, stream>>>(xb, wvb, bv, vb,
            M_ROWS, H_DIM, 1.0f, 0, 0, 0);
    }

    // 3) V transpose per batch: [S,H] -> [H,S]   (overwrites xb region)
    transpose_bf16_kernel<<<dim3(H_DIM / 64, S_LEN / 64, B_SZ), blk, 0, stream>>>(vb, vtb);

    // 4) scores = Q @ K^T / sqrt(H)   per batch (M=N=2048, K=1024), f32 out
    gemm_abt_kernel<false, H_DIM><<<dim3(S_LEN / 128, S_LEN / 128, B_SZ), blk, 0, stream>>>(
        qb, kb, nullptr, sc, S_LEN, S_LEN, 0.03125f,
        (long long)S_LEN * H_DIM, (long long)S_LEN * H_DIM,
        (long long)S_LEN * S_LEN);

    // 5) softmax rows -> bf16 P
    softmax_rows_kernel<<<dim3(M_ROWS / 8), blk, 0, stream>>>(sc, pb);

    // 6) attn_out = P @ V  ==  P @ (V^T)^T   per batch (M=2048, N=1024, K=2048)
    gemm_abt_kernel<true, S_LEN><<<dim3(H_DIM / 128, S_LEN / 128, B_SZ), blk, 0, stream>>>(
        pb, vtb, nullptr, ab, S_LEN, H_DIM, 1.0f,
        (long long)S_LEN * S_LEN, (long long)H_DIM * S_LEN,
        (long long)S_LEN * H_DIM);

    // 7) out = attn_out @ Wo^T + bo   (M=8192, N=K=1024), f32 out -> d_out
    gemm_abt_kernel<false, H_DIM><<<dim3(H_DIM / 128, M_ROWS / 128, 1), blk, 0, stream>>>(
        ab, wob, bo, out, M_ROWS, H_DIM, 1.0f, 0, 0, 0);
}
